// SACRSN_v55_23536420782597
// MI455X (gfx1250) — compile-verified, matrix operand order checked by reading, not run on hardware
//
#include <hip/hip_runtime.h>
#include <hip/hip_bf16.h>

// ---------------- types ----------------
typedef __bf16 bh;
typedef __bf16 v16bh __attribute__((ext_vector_type(16)));
typedef __bf16 v8bh  __attribute__((ext_vector_type(8)));
typedef float  v8f   __attribute__((ext_vector_type(8)));
typedef unsigned int u32x4 __attribute__((ext_vector_type(4)));
typedef int i32x4 __attribute__((ext_vector_type(4)));
typedef int i32x8 __attribute__((ext_vector_type(8)));

union Frag { v16bh v; v8bh h[2]; };

#define B_  4096
#define D_  256
#define S_  32
#define K_  8192
#define A_  256
#define TWOD 512
#define NCL 2560      // 5 clins * 512 outputs

__device__ __forceinline__ v8f zero8() {
  v8f z = {0.f,0.f,0.f,0.f,0.f,0.f,0.f,0.f};
  return z;
}

__device__ __forceinline__ v8f wmma_bf16(v16bh a, v16bh b, v8f c) {
  return __builtin_amdgcn_wmma_f32_16x16x32_bf16(false, a, false, b, (short)0, c, false, false);
}

__device__ __forceinline__ float wred_sum(float v) {
#pragma unroll
  for (int o = 16; o > 0; o >>= 1) v += __shfl_xor(v, o, 32);
  return v;
}
__device__ __forceinline__ float wred_max(float v) {
#pragma unroll
  for (int o = 16; o > 0; o >>= 1) v = fmaxf(v, __shfl_xor(v, o, 32));
  return v;
}
__device__ __forceinline__ float bred_sum(float v, float* tmp8) {
  v = wred_sum(v);
  if ((threadIdx.x & 31) == 0) tmp8[threadIdx.x >> 5] = v;
  __syncthreads();
  if (threadIdx.x == 0) {
    float s = 0.f;
#pragma unroll
    for (int i = 0; i < 8; ++i) s += tmp8[i];
    tmp8[0] = s;
  }
  __syncthreads();
  return tmp8[0];
}

// TDM: copy one contiguous 16KB block (16 rows x 512 bf16) global -> LDS.
// 1D transfer described as 2048 8-byte elements.
// Toolchain here is the clang-23 lane: 6-arg tensor_load_to_lds
// (g0 u32x4, g1 i32x8, g2 i32x4, g3 i32x4, extra i32x8, cpol i32).
__device__ __forceinline__ void tdm_copy_16k(const void* gsrc, void* lds_dst) {
#if __has_builtin(__builtin_amdgcn_tensor_load_to_lds) && __has_builtin(__builtin_amdgcn_s_wait_tensorcnt)
  unsigned long long ga = (unsigned long long)gsrc;
  unsigned lo = (unsigned)(unsigned long long)lds_dst;   // low 32 bits = LDS offset
  u32x4 g0 = { 1u,                                     // count=1
               lo,                                     // lds_addr
               (unsigned)ga,                           // global_addr[31:0]
               (unsigned)((ga >> 32) & 0x01FFFFFFu) | (2u << 30) }; // addr[56:32] | type=2
  i32x8 g1 = { (int)(3u << 16),        // data_size = 8B
               (int)(0x800u << 16),    // tensor_dim0 = 2048 (low16 at bit48)
               (int)(1u << 16),        // tensor_dim0 hi=0, tensor_dim1 = 1
               (int)(0x800u << 16),    // tile_dim0 = 2048
               1,                      // tile_dim1 = 1, tile_dim2 = 0
               2048,                   // tensor_dim0_stride
               (int)(0x800u << 16),    // tensor_dim1_stride = 2048
               0 };
  i32x4 g2 = {0,0,0,0};
  i32x4 g3 = {0,0,0,0};
  i32x8 g4 = {0,0,0,0,0,0,0,0};
  __builtin_amdgcn_tensor_load_to_lds(g0, g1, g2, g3, g4, 0);
  __builtin_amdgcn_s_wait_tensorcnt((short)0);
#else
  const uint4* s = (const uint4*)gsrc;
  uint4* d = (uint4*)lds_dst;
  for (int i = (threadIdx.x & 31); i < 1024; i += 32) d[i] = s[i];
#endif
}

// load a 16x32 bf16 WMMA A/B fragment from a row-major [*,512-or-256] slab,
// lane L<16 -> row m=L, K {0..7,16..23}; lane L>=16 -> row m=L-16, K {8..15,24..31}
__device__ __forceinline__ Frag load_frag(const bh* base, int row_stride, int row, int kbase) {
  Frag f;
  const bh* p = base + (size_t)row * row_stride + kbase;
  f.h[0] = *(const v8bh*)(p);
  f.h[1] = *(const v8bh*)(p + 16);
  return f;
}

// ---------------- kernel 0: input gate blend ----------------
__global__ void __launch_bounds__(256) prep_gate_kernel(
    const float* __restrict__ xr, const float* __restrict__ xi,
    const float* __restrict__ gwr, const float* __restrict__ gwi,
    const float* __restrict__ ig,
    float* __restrict__ gr, float* __restrict__ gi, bh* __restrict__ z2,
    float* __restrict__ sent) {
  int i = blockIdx.x * 256 + threadIdx.x;      // < B*D
  float alpha = 1.f / (1.f + __expf(-ig[0]));
  float r = alpha * gwr[i] + (1.f - alpha) * xr[i];
  float m = alpha * gwi[i] + (1.f - alpha) * xi[i];
  gr[i] = r;  gi[i] = m;
  int b = i >> 8, d = i & 255;
  z2[(size_t)b * TWOD + d] = (bh)r;
  z2[(size_t)b * TWOD + 256 + d] = (bh)m;
  if (i == 0) *sent = 0.f;
}

// ---------------- kernel 1: vq prep (bf16 cast + norms + zero hist) -------
__global__ void __launch_bounds__(256) vq_prep_kernel(
    const float* __restrict__ emb, bh* __restrict__ eb,
    float* __restrict__ en2, int* __restrict__ counts) {
  __shared__ float tmp8[8];
  int e = blockIdx.x;
  const float* src = emb + (size_t)e * TWOD;
  float acc = 0.f;
  for (int d = threadIdx.x; d < TWOD; d += 256) {
    float v = src[d];
    eb[(size_t)e * TWOD + d] = (bh)v;
    acc += v * v;
  }
  float s = bred_sum(acc, tmp8);
  if (threadIdx.x == 0) { en2[e] = s; counts[e] = 0; }
}

// ---------------- kernel 2: pack complex-linear weights ----------------
struct WPtrs {
  const float* Wr[5]; const float* Wi[5];
  const float* br[5]; const float* bi[5];
};

__global__ void __launch_bounds__(256) pack_w_kernel(WPtrs P, bh* __restrict__ Wcat,
                                                     float* __restrict__ bcat) {
  int r = blockIdx.x;                 // 0..2559
  int p = r / 512;
  int rr = r - p * 512;
  bool im = rr >= 256;
  int o = im ? rr - 256 : rr;
  const float* Wr = P.Wr[p] + (size_t)o * 256;
  const float* Wi = P.Wi[p] + (size_t)o * 256;
  int k = threadIdx.x;
  float w0, w1;
  if (!im) { w0 = Wr[k]; w1 = -Wi[k]; }
  else     { w0 = Wi[k]; w1 =  Wr[k]; }
  Wcat[(size_t)r * TWOD + k] = (bh)w0;
  Wcat[(size_t)r * TWOD + 256 + k] = (bh)w1;
  if (k == 0) bcat[r] = im ? (P.br[p][o] + P.bi[p][o]) : (P.br[p][o] - P.bi[p][o]);
}

// ---------------- kernel 3: pack palettes (bf16 + transpose) ------------
__global__ void __launch_bounds__(256) pack_pal_kernel(
    const float* __restrict__ vp, const float* __restrict__ ap,
    bh* __restrict__ vb, bh* __restrict__ ab,
    bh* __restrict__ vT, bh* __restrict__ aT) {
  int e = blockIdx.x & 255;
  bool aud = blockIdx.x >= 256;
  const float* src = (aud ? ap : vp) + (size_t)e * TWOD;
  bh* dst  = aud ? ab : vb;
  bh* dstT = aud ? aT : vT;
  for (int d = threadIdx.x; d < TWOD; d += 256) {
    float v = src[d];
    dst[(size_t)e * TWOD + d] = (bh)v;
    dstT[(size_t)d * 256 + e] = (bh)v;
  }
}

// ---------------- kernel 4: VQ distance GEMM + argmin (WMMA) ------------
__global__ void __launch_bounds__(256) vq_argmin_kernel(
    const bh* __restrict__ z2, const bh* __restrict__ eb,
    const float* __restrict__ en2, int* __restrict__ idx_out,
    int* __restrict__ counts) {
  __shared__ bh As[16 * TWOD];          // 16 KB
  __shared__ float red_d[8][16][16];
  __shared__ int   red_i[8][16][16];
  int blk = blockIdx.x;
  int w = threadIdx.x >> 5, lane = threadIdx.x & 31;
  int hlf = lane >> 4, mrow = lane & 15, ncol = lane & 15;

  if (w == 0) tdm_copy_16k(z2 + (size_t)blk * 16 * TWOD, As);
  __syncthreads();

  Frag a[16];
#pragma unroll
  for (int kk = 0; kk < 16; ++kk) a[kk] = load_frag(As, TWOD, mrow, kk * 32 + hlf * 8);

  float bestd[8];
  int bestn[8];
#pragma unroll
  for (int j = 0; j < 8; ++j) { bestd[j] = 3.0e38f; bestn[j] = 0; }

  for (int t = 0; t < 64; ++t) {
    int nbase = (w * 64 + t) * 16;
    const bh* ebp = eb + (size_t)(nbase + ncol) * TWOD;
    if (t + 1 < 64)
      __builtin_prefetch(eb + (size_t)(nbase + 16 + ncol) * TWOD, 0, 1);
    v8f c = zero8();
#pragma unroll
    for (int kk = 0; kk < 16; ++kk) {
      Frag bfr;
      int kb = kk * 32 + hlf * 8;
      bfr.h[0] = *(const v8bh*)(ebp + kb);
      bfr.h[1] = *(const v8bh*)(ebp + kb + 16);
      c = wmma_bf16(a[kk].v, bfr.v, c);
    }
    float e2 = en2[nbase + ncol];
#pragma unroll
    for (int j = 0; j < 8; ++j) {
      float d2 = e2 - 2.0f * c[j];
      if (d2 < bestd[j]) { bestd[j] = d2; bestn[j] = nbase + ncol; }
    }
  }
#pragma unroll
  for (int j = 0; j < 8; ++j) {
    int m = j + hlf * 8;
    red_d[w][m][ncol] = bestd[j];
    red_i[w][m][ncol] = bestn[j];
  }
  __syncthreads();
  if (threadIdx.x < 16) {
    int m = threadIdx.x;
    float bd = 3.0e38f; int bn = 0;
    for (int ww = 0; ww < 8; ++ww)
      for (int l = 0; l < 16; ++l) {
        float d = red_d[ww][m][l];
        if (d < bd) { bd = d; bn = red_i[ww][m][l]; }
      }
    idx_out[blk * 16 + m] = bn;
    atomicAdd(&counts[bn], 1);
  }
}

// ---------------- kernel 5: VQ finalize (v_out = z_q, vq_loss) ----------
__global__ void __launch_bounds__(256) vq_finalize_kernel(
    const float* __restrict__ emb, const int* __restrict__ idx,
    const float* __restrict__ gr, const float* __restrict__ gi,
    float* __restrict__ out_main, float* __restrict__ vql) {
  __shared__ float tmp8[8];
  int b = blockIdx.x;
  int n = idx[b];
  const float* zq = emb + (size_t)n * TWOD;
  float acc = 0.f;
  for (int c = threadIdx.x; c < TWOD; c += 256) {
    float q = zq[c];
    float z = (c < 256) ? gr[(size_t)b * 256 + c] : gi[(size_t)b * 256 + (c - 256)];
    out_main[(size_t)b * NCL + c] = q;      // slot 0: straight-through == z_q
    float d = q - z;
    acc += d * d;
  }
  float s = bred_sum(acc, tmp8);
  if (threadIdx.x == 0) vql[b] = 0.25f * s * (1.f / 512.f);
}

// ---------------- kernel 6: batch entropy ----------------
__global__ void __launch_bounds__(256) batch_ent_kernel(const int* __restrict__ counts,
                                                        float* __restrict__ bent) {
  __shared__ float tmp8[8];
  float acc = 0.f;
  for (int k = threadIdx.x; k < K_; k += 256) {
    float p = (float)counts[k] * (1.f / 4096.f);
    acc -= p * __logf(p + 1e-10f);
  }
  float s = bred_sum(acc, tmp8);
  if (threadIdx.x == 0) *bent = s / __logf(8192.f);
}

// ---------------- kernel 7: fused complex-linear GEMM (WMMA) ------------
__global__ void __launch_bounds__(256) clin_gemm_kernel(
    const bh* __restrict__ z2, const bh* __restrict__ Wcat,
    const float* __restrict__ bcat, float* __restrict__ cout) {
  __shared__ bh As[16 * TWOD];
  int blk = blockIdx.x;
  int w = threadIdx.x >> 5, lane = threadIdx.x & 31;
  int hlf = lane >> 4, mrow = lane & 15, ncol = lane & 15;

  if (w == 0) tdm_copy_16k(z2 + (size_t)blk * 16 * TWOD, As);
  __syncthreads();

  Frag a[16];
#pragma unroll
  for (int kk = 0; kk < 16; ++kk) a[kk] = load_frag(As, TWOD, mrow, kk * 32 + hlf * 8);

  for (int t = 0; t < 20; ++t) {
    int col = (w * 20 + t) * 16 + ncol;
    const bh* wp = Wcat + (size_t)col * TWOD;
    if (t + 1 < 20)
      __builtin_prefetch(Wcat + (size_t)(col + 16) * TWOD, 0, 1);
    v8f c = zero8();
#pragma unroll
    for (int kk = 0; kk < 16; ++kk) {
      Frag bfr;
      int kb = kk * 32 + hlf * 8;
      bfr.h[0] = *(const v8bh*)(wp + kb);
      bfr.h[1] = *(const v8bh*)(wp + kb + 16);
      c = wmma_bf16(a[kk].v, bfr.v, c);
    }
    float bias = bcat[col];
#pragma unroll
    for (int j = 0; j < 8; ++j) {
      int m = j + hlf * 8;
      cout[(size_t)(blk * 16 + m) * NCL + col] = c[j] + bias;
    }
  }
}

// ---------------- kernel 8: gating + critic + zf2 cast ----------------
__global__ void __launch_bounds__(256) gating_kernel(
    const float* __restrict__ cl, float* __restrict__ out_main, bh* __restrict__ zf2) {
  __shared__ float tmp8[8];
  int b = blockIdx.x, t = threadIdx.x;
  const float* row = cl + (size_t)b * NCL;
  float qr = row[t],        qi = row[256 + t];
  float kr = row[512 + t],  ki = row[768 + t];
  float vr = row[1024 + t], vi = row[1280 + t];
  float sr = row[1536 + t], si = row[1792 + t];
  float cr = row[2048 + t], ci = row[2304 + t];
  float dot = bred_sum(qr * kr + qi * ki, tmp8);
  float gate = 1.f / (1.f + __expf(-dot));
  float* om = out_main + (size_t)b * NCL;
  om[512 + t] = vr * gate;      // slot 1 real
  om[768 + t] = vi * gate;      // slot 1 imag
  om[2048 + t] = -ci;           // slot 4 real
  om[2304 + t] = cr;            // slot 4 imag
  zf2[(size_t)b * TWOD + t] = (bh)sr;
  zf2[(size_t)b * TWOD + 256 + t] = (bh)si;
}

// ---------------- kernel 9: palette attention (fused WMMA) --------------
__global__ void __launch_bounds__(256) palette_kernel(
    const bh* __restrict__ zf2, const bh* __restrict__ vb, const bh* __restrict__ ab,
    const bh* __restrict__ vT, const bh* __restrict__ aT,
    float* __restrict__ out_main) {
  extern __shared__ char sm[];
  bh* At = (bh*)sm;                       // 16 KB (reused as attn later)
  float* outb = (float*)(sm + 16384);     // 64 KB: [2][16][512]
  float* logit = outb;                    // [2][16][256] (first 32 KB)
  int blk = blockIdx.x, t = threadIdx.x;
  int w = t >> 5, lane = t & 31;
  int hlf = lane >> 4, mrow = lane & 15, ncol = lane & 15;

  // stage zf2 tile
  {
    const uint4* s = (const uint4*)(zf2 + (size_t)blk * 16 * TWOD);
    uint4* d = (uint4*)At;
#pragma unroll
    for (int k = 0; k < 4; ++k) d[t + k * 256] = s[t + k * 256];
  }
  __syncthreads();

  Frag a[16];
#pragma unroll
  for (int kk = 0; kk < 16; ++kk) a[kk] = load_frag(At, TWOD, mrow, kk * 32 + hlf * 8);

  // phase 1: logits = zf2 @ pal^T   (2 palettes x 16 col-tiles)
  for (int q = 0; q < 4; ++q) {
    int tt = w * 4 + q;
    int p = tt >> 4;
    int nb = (tt & 15) * 16;
    const bh* pal = (p ? ab : vb) + (size_t)(nb + ncol) * TWOD;
    v8f c = zero8();
#pragma unroll
    for (int kk = 0; kk < 16; ++kk) {
      Frag bfr;
      int kb = kk * 32 + hlf * 8;
      bfr.h[0] = *(const v8bh*)(pal + kb);
      bfr.h[1] = *(const v8bh*)(pal + kb + 16);
      c = wmma_bf16(a[kk].v, bfr.v, c);
    }
#pragma unroll
    for (int j = 0; j < 8; ++j) {
      int m = j + hlf * 8;
      logit[(p * 16 + m) * 256 + nb + ncol] = c[j];
    }
  }
  __syncthreads();

  // phase 2: softmax over 256 per (palette,row); write bf16 attn into At region
  bh* attn = At;                           // [2][16][256] = exactly 16 KB
  for (int q = 0; q < 4; ++q) {
    int r = w * 4 + q;                     // 0..31
    float v[8];
    float mx = -3.0e38f;
#pragma unroll
    for (int k = 0; k < 8; ++k) { v[k] = logit[r * 256 + lane + k * 32]; mx = fmaxf(mx, v[k]); }
    mx = wred_max(mx);
    float sum = 0.f;
#pragma unroll
    for (int k = 0; k < 8; ++k) { v[k] = __expf(v[k] - mx); sum += v[k]; }
    sum = wred_sum(sum);
    float inv = 1.f / sum;
#pragma unroll
    for (int k = 0; k < 8; ++k) attn[r * 256 + lane + k * 32] = (bh)(v[k] * inv);
  }
  __syncthreads();

  // phase 3: out = attn @ pal  (waves 0-3: vis, 4-7: aud; 8 col-tiles each)
  int p3 = w >> 2;
  Frag a2[8];
#pragma unroll
  for (int kk = 0; kk < 8; ++kk)
    a2[kk] = load_frag(attn + (size_t)p3 * 16 * 256, 256, mrow, kk * 32 + hlf * 8);
  const bh* palT = p3 ? aT : vT;
  for (int q = 0; q < 8; ++q) {
    int cb = ((w & 3) * 8 + q) * 16;
    const bh* bp = palT + (size_t)(cb + ncol) * 256;
    v8f c = zero8();
#pragma unroll
    for (int kk = 0; kk < 8; ++kk) {
      Frag bfr;
      int kb = kk * 32 + hlf * 8;
      bfr.h[0] = *(const v8bh*)(bp + kb);
      bfr.h[1] = *(const v8bh*)(bp + kb + 16);
      c = wmma_bf16(a2[kk].v, bfr.v, c);
    }
#pragma unroll
    for (int j = 0; j < 8; ++j) {
      int m = j + hlf * 8;
      outb[(p3 * 16 + m) * TWOD + cb + ncol] = c[j];
    }
  }
  __syncthreads();

  // phase 4: s_out = [vis[:, :256]-aud[:, 256:], vis[:, 256:]+aud[:, :256]]
  float* visb = outb;
  float* audb = outb + 16 * TWOD;
#pragma unroll 4
  for (int k = 0; k < 32; ++k) {
    int i = t + k * 256;                   // 0..8191
    int m = i >> 9, c = i & 511;
    float val = (c < 256) ? visb[m * TWOD + c] - audb[m * TWOD + 256 + c]
                          : visb[m * TWOD + c] + audb[m * TWOD + (c - 256)];
    out_main[(size_t)(blk * 16 + m) * NCL + 1536 + c] = val;  // slot 3
  }
}

// ---------------- kernel 10: associative memory (1 HBM pass) ------------
__global__ void __launch_bounds__(256) assoc_kernel(
    const float* __restrict__ mem_r, const float* __restrict__ mem_i,
    const float* __restrict__ gr, const float* __restrict__ gi,
    const float* __restrict__ mgW, const float* __restrict__ mgb,
    const float* __restrict__ maW, const float* __restrict__ mab,
    const float* __restrict__ lgr, const float* __restrict__ lbr,
    const float* __restrict__ lgi, const float* __restrict__ lbi,
    float* __restrict__ out_main, float* __restrict__ nm_r, float* __restrict__ nm_i,
    float* __restrict__ sent) {
  extern __shared__ char sm[];
  float* smr   = (float*)sm;                 // 32 KB [32][256]
  float* smi   = (float*)(sm + 32768);       // 32 KB
  float* readr = (float*)(sm + 65536);       // 1 KB
  float* readi = (float*)(sm + 66560);       // 1 KB
  float* simv  = (float*)(sm + 67584);       // 32
  float* wwl   = (float*)(sm + 67712);       // 32
  float* attnv = (float*)(sm + 67840);       // 32
  float* effv  = (float*)(sm + 67968);       // 32
  float* wgs   = (float*)(sm + 68096);       // 1

  int b = blockIdx.x, t = threadIdx.x;
  int w = t >> 5, lane = t & 31;

  // stage this row's memory (single HBM read)
  {
    const uint4* sr4 = (const uint4*)(mem_r + (size_t)b * S_ * D_);
    const uint4* si4 = (const uint4*)(mem_i + (size_t)b * S_ * D_);
    uint4* dr4 = (uint4*)smr;
    uint4* di4 = (uint4*)smi;
#pragma unroll
    for (int k = 0; k < 8; ++k) { dr4[t + k * 256] = sr4[t + k * 256]; di4[t + k * 256] = si4[t + k * 256]; }
  }
  readr[t] = 0.f; readi[t] = 0.f;
  __syncthreads();

  float grv[8], giv[8];
#pragma unroll
  for (int k = 0; k < 8; ++k) {
    int d = lane + k * 32;
    grv[k] = gr[(size_t)b * D_ + d];
    giv[k] = gi[(size_t)b * D_ + d];
  }

  // sim + addressing logits (wave handles 4 slots)
  for (int q = 0; q < 4; ++q) {
    int s = w * 4 + q;
    float a0 = 0.f, a1 = 0.f;
#pragma unroll
    for (int k = 0; k < 8; ++k) {
      int d = lane + k * 32;
      float mr = smr[s * D_ + d], mi = smi[s * D_ + d];
      a0 += mr * grv[k] + mi * giv[k];
      a1 += grv[k] * maW[(size_t)s * TWOD + d] + giv[k] * maW[(size_t)s * TWOD + 256 + d];
    }
    a0 = wred_sum(a0);
    a1 = wred_sum(a1);
    if (lane == 0) { simv[s] = a0; wwl[s] = a1 + mab[s]; }
  }
  if (w == 0) {
    float a = 0.f;
#pragma unroll
    for (int k = 0; k < 8; ++k) {
      int d = lane + k * 32;
      a += grv[k] * mgW[d] + giv[k] * mgW[256 + d];
    }
    a = wred_sum(a);
    if (lane == 0) wgs[0] = 1.f / (1.f + __expf(-(a + mgb[0])));
  }
  __syncthreads();

  // softmaxes over 32 slots (one lane per slot)
  if (w == 0) {
    float sv = simv[lane], lv = wwl[lane];
    float smax = wred_max(sv), lmax = wred_max(lv);
    float se = __expf(sv - smax), le = __expf(lv - lmax);
    float ssum = wred_sum(se), lsum = wred_sum(le);
    float at = se / ssum, wv = le / lsum;
    attnv[lane] = at;
    effv[lane] = wgs[0] * wv;
    float ent = wred_sum(-wv * __logf(wv + 1e-10f));
    if (lane == 0) atomicAdd(sent, ent * (1.f / 4096.f));
  }
  __syncthreads();

  // LN write + attention read accumulation (ds_add_f32 LDS atomics)
  for (int q = 0; q < 4; ++q) {
    int s = w * 4 + q;
    float e = effv[s], at = attnv[s];
    float vr[8], vi[8];
    float s1r = 0.f, s2r = 0.f, s1i = 0.f, s2i = 0.f;
#pragma unroll
    for (int k = 0; k < 8; ++k) {
      int d = lane + k * 32;
      float mr = smr[s * D_ + d], mi = smi[s * D_ + d];
      atomicAdd(&readr[d], at * mr);
      atomicAdd(&readi[d], at * mi);
      vr[k] = (1.f - e) * mr + e * grv[k];
      vi[k] = (1.f - e) * mi + e * giv[k];
      s1r += vr[k]; s2r += vr[k] * vr[k];
      s1i += vi[k]; s2i += vi[k] * vi[k];
    }
    s1r = wred_sum(s1r); s2r = wred_sum(s2r);
    s1i = wred_sum(s1i); s2i = wred_sum(s2i);
    float mr_ = s1r * (1.f / 256.f), vrr = s2r * (1.f / 256.f) - mr_ * mr_;
    float mi_ = s1i * (1.f / 256.f), vii = s2i * (1.f / 256.f) - mi_ * mi_;
    float rr = rsqrtf(vrr + 1e-5f), ri = rsqrtf(vii + 1e-5f);
#pragma unroll
    for (int k = 0; k < 8; ++k) {
      int d = lane + k * 32;
      size_t o = (size_t)b * (S_ * D_) + (size_t)s * D_ + d;
      nm_r[o] = (vr[k] - mr_) * rr * lgr[d] + lbr[d];
      nm_i[o] = (vi[k] - mi_) * ri * lgi[d] + lbi[d];
    }
  }
  __syncthreads();
  out_main[(size_t)b * NCL + 1024 + t] = readr[t];   // slot 2 real
  out_main[(size_t)b * NCL + 1280 + t] = readi[t];   // slot 2 imag
}

// ---------------- host launch ----------------
extern "C" void kernel_launch(void* const* d_in, const int* in_sizes, int n_in,
                              void* d_out, int out_size, void* d_ws, size_t ws_size,
                              hipStream_t stream) {
  (void)in_sizes; (void)n_in; (void)out_size; (void)ws_size;

  const float* x_t_r   = (const float*)d_in[0];
  const float* x_t_i   = (const float*)d_in[1];
  const float* gw_r    = (const float*)d_in[2];
  const float* gw_i    = (const float*)d_in[3];
  const float* mem_r   = (const float*)d_in[4];
  const float* mem_i   = (const float*)d_in[5];
  const float* in_gate = (const float*)d_in[6];
  const float* vq_emb  = (const float*)d_in[7];
  const float* vis_pal = (const float*)d_in[8];
  const float* aud_pal = (const float*)d_in[9];
  const float* mgW     = (const float*)d_in[10];
  const float* mgb     = (const float*)d_in[11];
  const float* maW     = (const float*)d_in[12];
  const float* mab     = (const float*)d_in[13];
  const float* mn_gr   = (const float*)d_in[14];
  const float* mn_br   = (const float*)d_in[15];
  const float* mn_gi   = (const float*)d_in[16];
  const float* mn_bi   = (const float*)d_in[17];

  WPtrs P;
  for (int p = 0; p < 5; ++p) {
    P.Wr[p] = (const float*)d_in[18 + 4 * p + 0];
    P.Wi[p] = (const float*)d_in[18 + 4 * p + 1];
    P.br[p] = (const float*)d_in[18 + 4 * p + 2];
    P.bi[p] = (const float*)d_in[18 + 4 * p + 3];
  }

  // workspace layout (bytes)
  char* ws = (char*)d_ws;
  float* gr       = (float*)(ws + 0);          // 4 MiB
  float* gi       = (float*)(ws + 4194304);    // 4 MiB
  bh*    z2       = (bh*)(ws + 8388608);       // 4 MiB
  bh*    eb       = (bh*)(ws + 12582912);      // 8 MiB
  float* en2      = (float*)(ws + 20971520);   // 32 KB
  int*   counts   = (int*)(ws + 21004288);     // 32 KB
  bh*    Wcat     = (bh*)(ws + 21037056);      // 2.5 MiB
  float* bcat     = (float*)(ws + 23658496);   // 10 KB
  float* clin_out = (float*)(ws + 23668736);   // 40 MiB
  bh*    zf2      = (bh*)(ws + 65611776);      // 4 MiB
  bh*    vpal_b   = (bh*)(ws + 69806080);
  bh*    apal_b   = (bh*)(ws + 70068224);
  bh*    vpal_T   = (bh*)(ws + 70330368);
  bh*    apal_T   = (bh*)(ws + 70592512);

  // output layout (elements)
  float* f_out    = (float*)d_out;
  float* out_main = f_out;                     // [B,5,512]
  float* nm_r     = f_out + 10485760;
  float* nm_i     = f_out + 44040192;
  int*   idx      = (int*)(f_out + 77594624);
  float* vql      = f_out + 77598720;
  float* bent     = f_out + 77602816;
  float* sent     = f_out + 77602817;

  prep_gate_kernel<<<4096, 256, 0, stream>>>(x_t_r, x_t_i, gw_r, gw_i, in_gate, gr, gi, z2, sent);
  vq_prep_kernel<<<8192, 256, 0, stream>>>(vq_emb, eb, en2, counts);
  pack_w_kernel<<<2560, 256, 0, stream>>>(P, Wcat, bcat);
  pack_pal_kernel<<<512, 256, 0, stream>>>(vis_pal, aud_pal, vpal_b, apal_b, vpal_T, apal_T);

  vq_argmin_kernel<<<256, 256, 0, stream>>>(z2, eb, en2, idx, counts);
  vq_finalize_kernel<<<4096, 256, 0, stream>>>(vq_emb, idx, gr, gi, out_main, vql);
  batch_ent_kernel<<<1, 256, 0, stream>>>(counts, bent);

  clin_gemm_kernel<<<256, 256, 0, stream>>>(z2, Wcat, bcat, clin_out);
  gating_kernel<<<4096, 256, 0, stream>>>(clin_out, out_main, zf2);
  palette_kernel<<<256, 256, 81920, stream>>>(zf2, vpal_b, apal_b, vpal_T, apal_T, out_main);

  assoc_kernel<<<4096, 256, 68608, stream>>>(mem_r, mem_i, gr, gi, mgW, mgb, maW, mab,
                                             mn_gr, mn_br, mn_gi, mn_bi,
                                             out_main, nm_r, nm_i, sent);
}
